// TDU_79534204387619
// MI455X (gfx1250) — matrix operand order repaired and checked by reading, not compiled, not run on hardware
//
#include <hip/hip_runtime.h>
#include <hip/hip_bf16.h>

#define CTROWS 8192
#define DDIM   1024
#define CROWS  65536

typedef __attribute__((ext_vector_type(16))) __bf16 v16bf;
typedef __attribute__((ext_vector_type(8)))  __bf16 v8bf;
typedef __attribute__((ext_vector_type(4)))  __bf16 v4bf;
typedef __attribute__((ext_vector_type(8)))  float  v8f;
typedef __attribute__((ext_vector_type(4)))  float  v4f;

// ---------------------------------------------------------------------------
// Elementwise split: x -> (hi = bf16(x), lo = bf16(x - hi)); vectorized x4
// ---------------------------------------------------------------------------
__global__ __launch_bounds__(256) void split_bf16_kernel(
    const float* __restrict__ x, __bf16* __restrict__ hi,
    __bf16* __restrict__ lo, int n4)
{
    int i = blockIdx.x * blockDim.x + threadIdx.x;
    if (i >= n4) return;
    v4f v = ((const v4f*)x)[i];
    v4bf h, l;
#pragma unroll
    for (int j = 0; j < 4; ++j) {
        __bf16 hb = (__bf16)v[j];
        h[j] = hb;
        l[j] = (__bf16)(v[j] - (float)hb);
    }
    ((v4bf*)hi)[i] = h;
    ((v4bf*)lo)[i] = l;
}

// ---------------------------------------------------------------------------
// Zero-fill (float4 stores) for the 256MB F_hat_t surface
// ---------------------------------------------------------------------------
__global__ __launch_bounds__(256) void zero_kernel(float* __restrict__ p, size_t n4)
{
    size_t i = (size_t)blockIdx.x * blockDim.x + threadIdx.x;
    if (i >= n4) return;
    v4f z = {0.f, 0.f, 0.f, 0.f};
    ((v4f*)p)[i] = z;
}

// ---------------------------------------------------------------------------
// Fragment loaders (wave32). Memory is row-major [row, k], ld = DDIM.
//
// A tile (16x32, 16-bit): lane = 16*h + m (h=0/1, m=0..15)
//   v0..v3 : K = h*8 .. h*8+7      (one b128)
//   v4..v7 : K = 16 + h*8 ..       (one b128)
// B tile (32x16, 16-bit), B[k,n] = W[n,k]: lane = 16*h + n
//   v0..v3 : K = h*8 .. h*8+7, v4..v7 : K = 16 + h*8 ..  (two b128)
// ---------------------------------------------------------------------------
__device__ __forceinline__ v16bf load_a_frag(const __bf16* __restrict__ base,
                                             int row0, int k0, int lane)
{
    const __bf16* q = base + (size_t)(row0 + (lane & 15)) * DDIM + k0 + ((lane >> 4) << 3);
    v8bf c0 = *(const v8bf*)(q);
    v8bf c1 = *(const v8bf*)(q + 16);
    v16bf a;
#pragma unroll
    for (int j = 0; j < 8; ++j) { a[j] = c0[j]; a[8 + j] = c1[j]; }
    return a;
}

__device__ __forceinline__ v16bf load_b_frag(const __bf16* __restrict__ base,
                                             int col0, int k0, int lane)
{
    const __bf16* q = base + (size_t)(col0 + (lane & 15)) * DDIM + k0 + ((lane >> 4) << 3);
    v8bf c0 = *(const v8bf*)(q);
    v8bf c1 = *(const v8bf*)(q + 16);
    v16bf b;
#pragma unroll
    for (int j = 0; j < 8; ++j) { b[j] = c0[j]; b[8 + j] = c1[j]; }
    return b;
}

__device__ __forceinline__ v8f wmma_bf16(v16bf a, v16bf b, v8f c)
{
    return __builtin_amdgcn_wmma_f32_16x16x32_bf16(
        false, a, false, b, (short)0, c, false, false);
}

// Split-bf16 product: C += Ah*Bh + Ah*Bl + Al*Bh  (~fp32 accuracy)
__device__ __forceinline__ v8f wmma_split(v16bf ah, v16bf al, v16bf bh, v16bf bl, v8f c)
{
    c = wmma_bf16(ah, bh, c);
    c = wmma_bf16(ah, bl, c);
    c = wmma_bf16(al, bh, c);
    return c;
}

__device__ __forceinline__ float sigmoidf_(float x)
{
    return 1.0f / (1.0f + __expf(-x));
}

// ---------------------------------------------------------------------------
// Kernel 2: fused R/Z GEMMs + sigmoid epilogue.
// Block = 128 threads (4 waves, 2x2), block tile 64x64, wave tile 32x32.
// Writes A2 = Ht*(1+R) as hi/lo bf16 splits, Z as fp32 into d_out F_t region.
// ---------------------------------------------------------------------------
__global__ __launch_bounds__(128) void gemm_rz_kernel(
    const __bf16* __restrict__ Ah,  const __bf16* __restrict__ Al,
    const __bf16* __restrict__ Wrh, const __bf16* __restrict__ Wrl,
    const __bf16* __restrict__ Wzh, const __bf16* __restrict__ Wzl,
    const float*  __restrict__ Ht,  const float* __restrict__ br,
    const float*  __restrict__ bz,
    __bf16* __restrict__ A2h, __bf16* __restrict__ A2l, float* __restrict__ Ztmp)
{
    const int lane = threadIdx.x & 31;
    const int wave = threadIdx.x >> 5;
    const int m0 = blockIdx.y * 64 + (wave >> 1) * 32;
    const int n0 = blockIdx.x * 64 + (wave & 1) * 32;

    v8f zero8 = {0.f, 0.f, 0.f, 0.f, 0.f, 0.f, 0.f, 0.f};
    v8f accR[2][2], accZ[2][2];
#pragma unroll
    for (int tm = 0; tm < 2; ++tm)
#pragma unroll
        for (int tn = 0; tn < 2; ++tn) { accR[tm][tn] = zero8; accZ[tm][tn] = zero8; }

    for (int k0 = 0; k0 < DDIM; k0 += 32) {
        v16bf ah[2], al[2];
#pragma unroll
        for (int tm = 0; tm < 2; ++tm) {
            ah[tm] = load_a_frag(Ah, m0 + 16 * tm, k0, lane);
            al[tm] = load_a_frag(Al, m0 + 16 * tm, k0, lane);
        }
#pragma unroll
        for (int tn = 0; tn < 2; ++tn) {
            v16bf brh = load_b_frag(Wrh, n0 + 16 * tn, k0, lane);
            v16bf brl = load_b_frag(Wrl, n0 + 16 * tn, k0, lane);
            v16bf bzh = load_b_frag(Wzh, n0 + 16 * tn, k0, lane);
            v16bf bzl = load_b_frag(Wzl, n0 + 16 * tn, k0, lane);
#pragma unroll
            for (int tm = 0; tm < 2; ++tm) {
                accR[tm][tn] = wmma_split(ah[tm], al[tm], brh, brl, accR[tm][tn]);
                accZ[tm][tn] = wmma_split(ah[tm], al[tm], bzh, bzl, accZ[tm][tn]);
            }
        }
    }

    // Epilogue.  C layout: vgpr v, lane l -> row = v + 8*(l>>4), col = l&15
#pragma unroll
    for (int tm = 0; tm < 2; ++tm)
#pragma unroll
        for (int tn = 0; tn < 2; ++tn) {
            int col = n0 + 16 * tn + (lane & 15);
            float vbr = br[col], vbz = bz[col];
#pragma unroll
            for (int v = 0; v < 8; ++v) {
                int row = m0 + 16 * tm + v + ((lane >> 4) << 3);
                size_t idx = (size_t)row * DDIM + col;
                float r = sigmoidf_(accR[tm][tn][v] + vbr);
                float z = sigmoidf_(accZ[tm][tn][v] + vbz);
                float h = Ht[idx];
                float a2 = h * (1.0f + r);           // R*Ht + Ht
                __bf16 a2h = (__bf16)a2;
                A2h[idx] = a2h;
                A2l[idx] = (__bf16)(a2 - (float)a2h);
                Ztmp[idx] = z;
            }
        }
}

// ---------------------------------------------------------------------------
// Kernel 3: F_tilde GEMM + tanh + gate blend + GI blend + row scatter.
// ZFt region: reads Z, overwrites with F_t (same element, same thread).
// ---------------------------------------------------------------------------
__global__ __launch_bounds__(128) void gemm_f_kernel(
    const __bf16* __restrict__ A2h, const __bf16* __restrict__ A2l,
    const __bf16* __restrict__ Wfh, const __bf16* __restrict__ Wfl,
    const float*  __restrict__ Ht,  const float* __restrict__ bf,
    const int*    __restrict__ code_rows,
    const float*  __restrict__ GI,  const float* __restrict__ Fhat_prev,
    float* __restrict__ Fhat_out,   float* ZFt)
{
    const int lane = threadIdx.x & 31;
    const int wave = threadIdx.x >> 5;
    const int m0 = blockIdx.y * 64 + (wave >> 1) * 32;
    const int n0 = blockIdx.x * 64 + (wave & 1) * 32;

    v8f zero8 = {0.f, 0.f, 0.f, 0.f, 0.f, 0.f, 0.f, 0.f};
    v8f acc[2][2];
#pragma unroll
    for (int tm = 0; tm < 2; ++tm)
#pragma unroll
        for (int tn = 0; tn < 2; ++tn) acc[tm][tn] = zero8;

    for (int k0 = 0; k0 < DDIM; k0 += 32) {
        v16bf ah[2], al[2];
#pragma unroll
        for (int tm = 0; tm < 2; ++tm) {
            ah[tm] = load_a_frag(A2h, m0 + 16 * tm, k0, lane);
            al[tm] = load_a_frag(A2l, m0 + 16 * tm, k0, lane);
        }
#pragma unroll
        for (int tn = 0; tn < 2; ++tn) {
            v16bf bh = load_b_frag(Wfh, n0 + 16 * tn, k0, lane);
            v16bf bl = load_b_frag(Wfl, n0 + 16 * tn, k0, lane);
#pragma unroll
            for (int tm = 0; tm < 2; ++tm)
                acc[tm][tn] = wmma_split(ah[tm], al[tm], bh, bl, acc[tm][tn]);
        }
    }

#pragma unroll
    for (int tm = 0; tm < 2; ++tm)
#pragma unroll
        for (int tn = 0; tn < 2; ++tn) {
            int col = n0 + 16 * tn + (lane & 15);
            float vbf = bf[col];
#pragma unroll
            for (int v = 0; v < 8; ++v) {
                int row = m0 + 16 * tm + v + ((lane >> 4) << 3);
                size_t idx = (size_t)row * DDIM + col;
                float ftld = tanhf(acc[tm][tn][v] + vbf);
                float h = Ht[idx];
                float z = ZFt[idx];
                float f = (1.0f - z) * h + z * ftld;   // F_t
                ZFt[idx] = f;
                int cr = code_rows[row];
                size_t oidx = (size_t)cr * DDIM + col;
                float gi = GI[oidx];
                Fhat_out[oidx] = gi * Fhat_prev[oidx] + (1.0f - gi) * f;
            }
        }
}

// ---------------------------------------------------------------------------
extern "C" void kernel_launch(void* const* d_in, const int* in_sizes, int n_in,
                              void* d_out, int out_size, void* d_ws, size_t ws_size,
                              hipStream_t stream)
{
    const float* Ht        = (const float*)d_in[0];
    /* d_in[1] = Ft_1 (unused by the forward body) */
    const float* Fhat_prev = (const float*)d_in[2];
    const int*   code_rows = (const int*)d_in[3];
    const float* W_r = (const float*)d_in[4];
    const float* b_r = (const float*)d_in[5];
    const float* W_z = (const float*)d_in[6];
    const float* b_z = (const float*)d_in[7];
    const float* W_f = (const float*)d_in[8];
    const float* b_f = (const float*)d_in[9];
    const float* GI  = (const float*)d_in[10];

    float* Fhat_out = (float*)d_out;                          // [C, D]
    float* Ft_out   = (float*)d_out + (size_t)CROWS * DDIM;   // [CT, D]  (also Z temp)

    // Workspace layout (bf16 splits): 76 MiB total
    char* ws = (char*)d_ws;
    size_t o = 0;
    const size_t szH = (size_t)CTROWS * DDIM * sizeof(__bf16);   // 16 MiB
    const size_t szW = (size_t)DDIM * DDIM * sizeof(__bf16);     //  2 MiB
    __bf16* Hh  = (__bf16*)(ws + o); o += szH;
    __bf16* Hl  = (__bf16*)(ws + o); o += szH;
    __bf16* Wrh = (__bf16*)(ws + o); o += szW;
    __bf16* Wrl = (__bf16*)(ws + o); o += szW;
    __bf16* Wzh = (__bf16*)(ws + o); o += szW;
    __bf16* Wzl = (__bf16*)(ws + o); o += szW;
    __bf16* Wfh = (__bf16*)(ws + o); o += szW;
    __bf16* Wfl = (__bf16*)(ws + o); o += szW;
    __bf16* A2h = (__bf16*)(ws + o); o += szH;
    __bf16* A2l = (__bf16*)(ws + o); o += szH;

    // 1) split fp32 -> hi/lo bf16
    int nH4 = CTROWS * DDIM / 4;
    int nW4 = DDIM * DDIM / 4;
    split_bf16_kernel<<<(nH4 + 255) / 256, 256, 0, stream>>>(Ht,  Hh,  Hl,  nH4);
    split_bf16_kernel<<<(nW4 + 255) / 256, 256, 0, stream>>>(W_r, Wrh, Wrl, nW4);
    split_bf16_kernel<<<(nW4 + 255) / 256, 256, 0, stream>>>(W_z, Wzh, Wzl, nW4);
    split_bf16_kernel<<<(nW4 + 255) / 256, 256, 0, stream>>>(W_f, Wfh, Wfl, nW4);

    // 2) zero F_hat_t (rows not hit by the scatter must be 0; scatter rows
    //    are overwritten afterwards — stream order guarantees this)
    size_t nZ4 = (size_t)CROWS * DDIM / 4;
    zero_kernel<<<(unsigned)((nZ4 + 255) / 256), 256, 0, stream>>>(Fhat_out, nZ4);

    // 3) fused R/Z GEMMs (+ sigmoid, A2 split, Z stash)
    dim3 grid(DDIM / 64, CTROWS / 64);
    gemm_rz_kernel<<<grid, 128, 0, stream>>>(Hh, Hl, Wrh, Wrl, Wzh, Wzl,
                                             Ht, b_r, b_z, A2h, A2l, Ft_out);

    // 4) F GEMM (+ tanh, gate blend, GI blend, scatter)
    gemm_f_kernel<<<grid, 128, 0, stream>>>(A2h, A2l, Wfh, Wfl, Ht, b_f,
                                            code_rows, GI, Fhat_prev,
                                            Fhat_out, Ft_out);
}
